// XTransformer_26190710571111
// MI455X (gfx1250) — matrix operand order repaired and checked by reading, not compiled, not run on hardware
//
#include <hip/hip_runtime.h>
#include <hip/hip_bf16.h>

typedef __attribute__((ext_vector_type(16))) _Float16 v16h;
typedef __attribute__((ext_vector_type(8)))  _Float16 v8h;
typedef __attribute__((ext_vector_type(8)))  float    v8f;

union V16 { v16h v; v8h h[2]; };

constexpr int TS  = 4096;   // sequence length S
constexpr int TD  = 128;    // model dim D
constexpr int TH  = 4;      // heads
constexpr int TDH = 32;     // head dim
constexpr int TL  = 3;      // layers
constexpr int TFF = 1024;   // ffn dim

// ---------------------------------------------------------------------------
// f32 -> f16 conversion (weights once per call, activations once per LN)
// ---------------------------------------------------------------------------
__global__ __launch_bounds__(256) void cvt_f32_f16_kernel(
    const float* __restrict__ in, _Float16* __restrict__ out, int n)
{
  const int i = blockIdx.x * 256 + threadIdx.x;
  if (i < n) out[i] = (_Float16)in[i];
}

// ---------------------------------------------------------------------------
// WMMA GEMM, f16 operands, f32 accumulate: C[MxN] = A[MxK] @ W[NxK]^T + bias.
// One 16x64 C tile per wave. Ping-pong software pipeline (requires K % 64
// == 0): loads write directly into the idle frag buffer's registers while
// the other buffer feeds 4 WMMAs -> no inter-buffer register copies.
// ep: 0 = none, 1 = relu, 2 = clamp >= 1e-6.  Output f32 (Cf) or f16 (Ch).
// ---------------------------------------------------------------------------
__global__ __launch_bounds__(256) void gemm16_wmma_kernel(
    const _Float16* __restrict__ A, const _Float16* __restrict__ W,
    const float* __restrict__ bias,
    float* __restrict__ Cf, _Float16* __restrict__ Ch,
    int M, int N, int K, int ep)
{
  const int lane = threadIdx.x & 31;
  const int wave = threadIdx.x >> 5;
  const int tN   = N >> 6;                 // 64-wide column tiles
  const int total = (M >> 4) * tN;
  const int t = blockIdx.x * 8 + wave;
  if (t >= total) return;                  // wave-uniform (grids are exact)

  const int tm  = t / tN;
  const int tn  = t - tm * tN;
  const int hi  = lane >> 4;
  const int l16 = lane & 15;
  const int ka  = hi ? 8  : 0;             // A-frag K base for this half
  const int kbb = hi ? 8 : 0;              // B-frag K base for this half

  const _Float16* ap0 = A + (size_t)(tm * 16 + l16) * K + ka;
  const _Float16* wp0 = W + (size_t)(tn * 64 + l16) * K + kbb;
  const size_t wstep = (size_t)16 * K;     // between 16-col sub-tiles

  auto lda = [&](int k0, V16& a) {
    a.h[0] = *(const v8h*)(ap0 + k0);          // K = ka..ka+7
    a.h[1] = *(const v8h*)(ap0 + k0 + 16);     // K = 16+ka..23+ka
  };
  auto ldb = [&](int k0, V16* b) {
    #pragma unroll
    for (int j = 0; j < 4; ++j) {
      const _Float16* wp = wp0 + j * wstep + k0;
      b[j].h[0] = *(const v8h*)(wp);           // K = kbb..kbb+7
      b[j].h[1] = *(const v8h*)(wp + 16);      // K = 16+kbb..23+kbb
    }
  };

  v8f acc[4] = {};
  V16 a0, b0[4], a1, b1[4];
  lda(0, a0);  ldb(0, b0);
  lda(32, a1); ldb(32, b1);
  for (int k0 = 0;; k0 += 64) {
    #pragma unroll
    for (int j = 0; j < 4; ++j) {
      acc[j] = __builtin_amdgcn_wmma_f32_16x16x32_f16(
          false, a0.v, false, b0[j].v, (short)0, acc[j], false, false);
    }
    const int k2 = k0 + 64;
    const bool more = (k2 < K);              // K % 64 == 0: single condition
    if (more) { lda(k2, a0); ldb(k2, b0); }
    #pragma unroll
    for (int j = 0; j < 4; ++j) {
      acc[j] = __builtin_amdgcn_wmma_f32_16x16x32_f16(
          false, a1.v, false, b1[j].v, (short)0, acc[j], false, false);
    }
    if (!more) break;
    lda(k2 + 32, a1); ldb(k2 + 32, b1);
  }

  #pragma unroll
  for (int j = 0; j < 4; ++j) {
    const int col = tn * 64 + j * 16 + l16;
    const float bv = bias ? bias[col] : 0.0f;
    #pragma unroll
    for (int r = 0; r < 8; ++r) {
      const int row = tm * 16 + r + 8 * hi;
      float v = acc[j][r] + bv;
      if (ep == 1) v = fmaxf(v, 0.0f);
      if (ep == 2) v = fmaxf(v, 1.0e-6f);
      if (Ch) Ch[(size_t)row * N + col] = (_Float16)v;
      else    Cf[(size_t)row * N + col] = v;
    }
  }
}

// ---------------------------------------------------------------------------
// Flash attention: one wave per (head, 16-query tile). qkv is f16,
// rows [q(128) | k(128) | v(128)], stride 384 halves. Output f16.
// All LDS staging is wave-private: no block barriers in the 256-iter loop.
// ---------------------------------------------------------------------------
__global__ __launch_bounds__(256) void attn_flash_kernel(
    const _Float16* __restrict__ qkv, _Float16* __restrict__ out)
{
  __shared__ _Float16 pls[8][16 * 16];     // per-wave P staging (C->A layout)
  __shared__ _Float16 vls[8][16 * 32];     // per-wave V tile [key][dim]

  const int lane = threadIdx.x & 31;
  const int wave = threadIdx.x >> 5;
  const int gid  = blockIdx.x * 8 + wave;  // grid exact: no guard needed
  const int h    = gid & (TH - 1);
  const int qt   = gid >> 2;
  const int hi   = lane >> 4;
  const int l16  = lane & 15;
  const int ka   = hi ? 8  : 0;
  const int kbb  = hi ? 8 : 0;
  const float scale = 0.17677669529663687f;   // 1/sqrt(32)

  // Q A-frag (16x32): two contiguous b128 loads
  V16 aq;
  {
    const _Float16* qrow = qkv + (size_t)(qt * 16 + l16) * 384 + h * TDH;
    aq.h[0] = *(const v8h*)(qrow + ka);
    aq.h[1] = *(const v8h*)(qrow + 16 + ka);
  }

  v8f o0 = {}, o1 = {};
  float mrow[8], lrow[8];
  #pragma unroll
  for (int r = 0; r < 8; ++r) { mrow[r] = -3.0e38f; lrow[r] = 0.0f; }

  _Float16* myp = &pls[wave][0];
  _Float16* myv = &vls[wave][0];

  for (int t = 0; t < TS / 16; ++t) {
    // K tile B-frag: col = key, K = head dim; two 8-half chunks (kbb, 16+kbb)
    V16 bk;
    {
      const _Float16* krow = qkv + (size_t)(t * 16 + l16) * 384 + TD + h * TDH + kbb;
      bk.h[0] = *(const v8h*)(krow);
      bk.h[1] = *(const v8h*)(krow + 16);
    }
    // cooperative V-tile stage: lane (row=l16, chunk=hi) -> 2 b128 + 2 ds_b128
    {
      const _Float16* vsrc = qkv + (size_t)(t * 16 + l16) * 384 + 2 * TD
                             + h * TDH + hi * 16;
      *(v8h*)(myv + l16 * 32 + hi * 16)     = *(const v8h*)(vsrc);
      *(v8h*)(myv + l16 * 32 + hi * 16 + 8) = *(const v8h*)(vsrc + 8);
    }

    v8f zero = {};
    v8f s = __builtin_amdgcn_wmma_f32_16x16x32_f16(
        false, aq.v, false, bk.v, (short)0, zero, false, false);

    // online softmax; each 16-lane half owns rows (r + 8*hi)
    #pragma unroll
    for (int r = 0; r < 8; ++r) {
      const float sv = s[r] * scale;
      float mx = sv;
      mx = fmaxf(mx, __shfl_xor(mx, 1, 32));
      mx = fmaxf(mx, __shfl_xor(mx, 2, 32));
      mx = fmaxf(mx, __shfl_xor(mx, 4, 32));
      mx = fmaxf(mx, __shfl_xor(mx, 8, 32));
      const float mnew = fmaxf(mrow[r], mx);
      const float p = __expf(sv - mnew);
      float ps = p;
      ps += __shfl_xor(ps, 1, 32);
      ps += __shfl_xor(ps, 2, 32);
      ps += __shfl_xor(ps, 4, 32);
      ps += __shfl_xor(ps, 8, 32);
      const float alpha = __expf(mrow[r] - mnew);
      lrow[r] = lrow[r] * alpha + ps;
      o0[r] *= alpha;
      o1[r] *= alpha;
      mrow[r] = mnew;
      myp[(r + 8 * hi) * 16 + l16] = (_Float16)p;
    }
    __builtin_amdgcn_wave_barrier();       // wave-private LDS: order ds ops

    // P as A-frag: M=16 rows, K=16 valid keys, upper K half zero
    v16h ap;
    #pragma unroll
    for (int i = 0; i < 4; ++i) {
      ap[2 * i]     = myp[l16 * 16 + ka + 2 * i];
      ap[2 * i + 1] = myp[l16 * 16 + ka + 2 * i + 1];
    }
    #pragma unroll
    for (int i = 4; i < 8; ++i) {
      ap[2 * i] = (_Float16)0.0f; ap[2 * i + 1] = (_Float16)0.0f;
    }

    // V tile B-frags from LDS: K = key (16 valid, elements 8-15 zero padding)
    v16h bv0, bv1;
    #pragma unroll
    for (int v = 0; v < 8; ++v) {
      _Float16 e0 = (_Float16)0.0f, e1 = e0, f0 = e0, f1 = e0;
      if (v < 4) {
        e0 = myv[(ka + 2 * v)     * 32 + l16];
        e1 = myv[(ka + 2 * v + 1) * 32 + l16];
        f0 = myv[(ka + 2 * v)     * 32 + 16 + l16];
        f1 = myv[(ka + 2 * v + 1) * 32 + 16 + l16];
      }
      bv0[2 * v] = e0; bv0[2 * v + 1] = e1;
      bv1[2 * v] = f0; bv1[2 * v + 1] = f1;
    }
    __builtin_amdgcn_wave_barrier();

    o0 = __builtin_amdgcn_wmma_f32_16x16x32_f16(false, ap, false, bv0,
                                                (short)0, o0, false, false);
    o1 = __builtin_amdgcn_wmma_f32_16x16x32_f16(false, ap, false, bv1,
                                                (short)0, o1, false, false);
  }

  #pragma unroll
  for (int r = 0; r < 8; ++r) {
    const int row = qt * 16 + r + 8 * hi;
    const float inv = 1.0f / lrow[r];
    out[(size_t)row * TD + h * TDH + l16]      = (_Float16)(o0[r] * inv);
    out[(size_t)row * TD + h * TDH + 16 + l16] = (_Float16)(o1[r] * inv);
  }
}

// ---------------------------------------------------------------------------
// x = LayerNorm(x + o) * g + b, one 128-thread block per row (f32)
// ---------------------------------------------------------------------------
__global__ __launch_bounds__(128) void ln_residual_kernel(
    float* __restrict__ x, const float* __restrict__ o,
    const float* __restrict__ g, const float* __restrict__ b)
{
  __shared__ float red[4];
  const int row = blockIdx.x, tid = threadIdx.x;
  const float v = x[(size_t)row * TD + tid] + o[(size_t)row * TD + tid];

  float s = v;
  for (int m = 16; m >= 1; m >>= 1) s += __shfl_xor(s, m, 32);
  if ((tid & 31) == 0) red[tid >> 5] = s;
  __syncthreads();
  const float mean = (red[0] + red[1] + red[2] + red[3]) * (1.0f / TD);
  __syncthreads();

  const float d = v - mean;
  float q = d * d;
  for (int m = 16; m >= 1; m >>= 1) q += __shfl_xor(q, m, 32);
  if ((tid & 31) == 0) red[tid >> 5] = q;
  __syncthreads();
  const float var = (red[0] + red[1] + red[2] + red[3]) * (1.0f / TD);

  x[(size_t)row * TD + tid] = d * rsqrtf(var + 1e-5f) * g[tid] + b[tid];
}

// ---------------------------------------------------------------------------
// xn[row] = in[row]/||in[row]|| written as f16 for the Gram GEMM
// ---------------------------------------------------------------------------
__global__ __launch_bounds__(128) void rownorm_kernel(
    const float* __restrict__ in, _Float16* __restrict__ outh)
{
  __shared__ float red[4];
  const int row = blockIdx.x, tid = threadIdx.x;
  const float v = in[(size_t)row * TD + tid];
  float s = v * v;
  for (int m = 16; m >= 1; m >>= 1) s += __shfl_xor(s, m, 32);
  if ((tid & 31) == 0) red[tid >> 5] = s;
  __syncthreads();
  const float n = red[0] + red[1] + red[2] + red[3];
  outh[(size_t)row * TD + tid] = (_Float16)(v / sqrtf(n));
}

// ---------------------------------------------------------------------------
extern "C" void kernel_launch(void* const* d_in, const int* in_sizes, int n_in,
                              void* d_out, int out_size, void* d_ws, size_t ws_size,
                              hipStream_t stream)
{
  (void)in_sizes; (void)n_in; (void)out_size; (void)ws_size;
  const float* src  = (const float*)d_in[0];
  const float* Wqkv = (const float*)d_in[1];
  const float* bqkv = (const float*)d_in[2];
  const float* Wo   = (const float*)d_in[3];
  const float* bo   = (const float*)d_in[4];
  const float* ln1g = (const float*)d_in[5];
  const float* ln1b = (const float*)d_in[6];
  const float* W1   = (const float*)d_in[7];
  const float* b1   = (const float*)d_in[8];
  const float* W2   = (const float*)d_in[9];
  const float* b2   = (const float*)d_in[10];
  const float* ln2g = (const float*)d_in[11];
  const float* ln2b = (const float*)d_in[12];
  const float* fc1W = (const float*)d_in[13];
  const float* fc1b = (const float*)d_in[14];
  const float* fc2W = (const float*)d_in[15];
  const float* fc2b = (const float*)d_in[16];
  float* out = (float*)d_out;

  // f32 scratch
  float* xbuf = (float*)d_ws;                        // S*D
  float* tbuf = xbuf + (size_t)TS * TD;              // S*D
  // f16 scratch
  _Float16* hws   = (_Float16*)(tbuf + (size_t)TS * TD);
  _Float16* xh    = hws;                             // S*D
  _Float16* qkvh  = xh   + (size_t)TS * TD;          // S*3D
  _Float16* oh    = qkvh + (size_t)TS * 3 * TD;      // S*D
  _Float16* ffh   = oh   + (size_t)TS * TD;          // S*FF
  _Float16* xnh   = ffh  + (size_t)TS * TFF;         // S*D
  _Float16* Wqkvh = xnh  + (size_t)TS * TD;          // L*3D*D
  _Float16* Woh   = Wqkvh + (size_t)TL * 3 * TD * TD;
  _Float16* W1h   = Woh   + (size_t)TL * TD * TD;
  _Float16* W2h   = W1h   + (size_t)TL * TFF * TD;
  _Float16* fc1h  = W2h   + (size_t)TL * TD * TFF;
  _Float16* fc2h  = fc1h  + (size_t)TD * TD;

  auto cvt = [&](const float* s, _Float16* d, size_t n) {
    cvt_f32_f16_kernel<<<(int)((n + 255) / 256), 256, 0, stream>>>(s, d, (int)n);
  };
  auto gemm = [&](const _Float16* A, const _Float16* W, const float* bias,
                  float* Cf, _Float16* Ch, int M, int N, int K, int ep) {
    const int total = (M / 16) * (N / 64);
    gemm16_wmma_kernel<<<(total + 7) / 8, 256, 0, stream>>>(A, W, bias, Cf, Ch,
                                                            M, N, K, ep);
  };

  // convert all weights to f16 (L2-resident, trivial cost)
  cvt(Wqkv, Wqkvh, (size_t)TL * 3 * TD * TD);
  cvt(Wo,   Woh,   (size_t)TL * TD * TD);
  cvt(W1,   W1h,   (size_t)TL * TFF * TD);
  cvt(W2,   W2h,   (size_t)TL * TD * TFF);
  cvt(fc1W, fc1h,  (size_t)TD * TD);
  cvt(fc2W, fc2h,  (size_t)TD * TD);

  hipMemcpyAsync(xbuf, src, (size_t)TS * TD * sizeof(float),
                 hipMemcpyDeviceToDevice, stream);
  cvt(xbuf, xh, (size_t)TS * TD);

  for (int l = 0; l < TL; ++l) {
    gemm(xh, Wqkvh + (size_t)l * 3 * TD * TD, bqkv + l * 3 * TD,
         nullptr, qkvh, TS, 3 * TD, TD, 0);
    attn_flash_kernel<<<(TH * (TS / 16)) / 8, 256, 0, stream>>>(qkvh, oh);
    gemm(oh, Woh + (size_t)l * TD * TD, bo + l * TD, tbuf, nullptr, TS, TD, TD, 0);
    ln_residual_kernel<<<TS, 128, 0, stream>>>(xbuf, tbuf, ln1g + l * TD, ln1b + l * TD);
    cvt(xbuf, xh, (size_t)TS * TD);
    gemm(xh, W1h + (size_t)l * TFF * TD, b1 + l * TFF, nullptr, ffh, TS, TFF, TD, 1);
    gemm(ffh, W2h + (size_t)l * TD * TFF, b2 + l * TD, tbuf, nullptr, TS, TD, TFF, 0);
    ln_residual_kernel<<<TS, 128, 0, stream>>>(xbuf, tbuf, ln2g + l * TD, ln2b + l * TD);
    cvt(xbuf, xh, (size_t)TS * TD);
  }

  gemm(xh, fc1h, fc1b, nullptr, oh, TS, TD, TD, 1);
  gemm(oh, fc2h, fc2b, tbuf, nullptr, TS, TD, TD, 0);
  rownorm_kernel<<<TS, 128, 0, stream>>>(tbuf, xnh);
  // Gram: out = clamp(xn @ xn^T, 1e-6), f16 operands, f32 out
  gemm(xnh, xnh, nullptr, out, nullptr, TS, TS, TD, 2);
}
